// NodeClassifier_66030827209234
// MI455X (gfx1250) — compile-verified, matrix operand order checked by reading, not run on hardware
//
#include <hip/hip_runtime.h>

typedef __attribute__((ext_vector_type(16))) _Float16 v16h;
typedef __attribute__((ext_vector_type(8)))  _Float16 v8h;
typedef __attribute__((ext_vector_type(8)))  float    v8f;

#define DIM 256

// ---------------- elementwise helpers ----------------

__global__ __launch_bounds__(256) void k_zero(float* __restrict__ p, int n) {
  int i = blockIdx.x * 256 + threadIdx.x;
  if (i < n) p[i] = 0.0f;
}

__global__ __launch_bounds__(256) void k_cvt_f16(const float* __restrict__ in,
                                                 _Float16* __restrict__ out, int n) {
  int i = blockIdx.x * 256 + threadIdx.x;
  if (i < n) out[i] = (_Float16)in[i];
}

__global__ __launch_bounds__(256) void k_lrelu_cvt(const float* __restrict__ in,
                                                   _Float16* __restrict__ out, int n) {
  int i = blockIdx.x * 256 + threadIdx.x;
  if (i < n) {
    float v = in[i];
    v = (v > 0.0f) ? v : 0.01f * v;
    out[i] = (_Float16)v;
  }
}

__global__ __launch_bounds__(256) void k_hist(const int* __restrict__ dst,
                                              int* __restrict__ cnt, int n) {
  int i = blockIdx.x * 256 + threadIdx.x;
  if (i < n) atomicAdd(&cnt[dst[i]], 1);
}

// in-place: buffer holds int counts, overwritten with float reciprocals
__global__ __launch_bounds__(256) void k_invert(float* __restrict__ inv, int n) {
  int i = blockIdx.x * 256 + threadIdx.x;
  if (i < n) {
    int c = ((const int*)inv)[i];
    inv[i] = 1.0f / (float)(c > 0 ? c : 1);
  }
}

// W [K=256][N=256] f32 row-major -> Wt [N=256][K=256] f16 (column-major B feed)
__global__ __launch_bounds__(256) void k_wtrans(const float* __restrict__ W,
                                                _Float16* __restrict__ Wt) {
  int i = blockIdx.x * 256 + threadIdx.x;   // 65536 threads
  int n = i >> 8, k = i & 255;
  Wt[n * DIM + k] = (_Float16)W[k * DIM + n];
}

// ---------------- WMMA GEMM: C[M,256] = A_f16[M,256] @ Wt_f16^T + bias ----------------
// One wave per block computes a 32x64 output tile (2 row-strips x 4 col-tiles):
// each B fragment is reused by two independent WMMA chains, halving B traffic
// and giving the scheduler independent work to hide global_load latency.
// grid.x = ceil(M/32), grid.y = 4 (64 cols per block), blockDim = 32 (one wave)

__device__ __forceinline__ v16h ldfrag(const _Float16* base, int ks, int kb) {
  v8h lo = *(const v8h*)(base + ks + kb);
  v8h hi = *(const v8h*)(base + ks + 16 + kb);
  return __builtin_shufflevector(lo, hi, 0,1,2,3,4,5,6,7,8,9,10,11,12,13,14,15);
}

__global__ __launch_bounds__(32) void k_gemm_wmma(const _Float16* __restrict__ A,
                                                  const _Float16* __restrict__ Wt,
                                                  const float* __restrict__ bias,
                                                  float* __restrict__ C, int M) {
  const int l  = threadIdx.x;
  const int m0 = blockIdx.x * 32;
  const int n0 = blockIdx.y * 64;
  const int lc = l & 15;
  const int kb = (l >> 4) * 8;     // K sub-chunk AND row offset for C layout

  int row0 = m0 + lc;       if (row0 >= M) row0 = M - 1;   // clamp loads
  int row1 = m0 + 16 + lc;  if (row1 >= M) row1 = M - 1;
  const _Float16* arow0 = A + (size_t)row0 * DIM;
  const _Float16* arow1 = A + (size_t)row1 * DIM;

  const _Float16* bp[4];
#pragma unroll
  for (int nt = 0; nt < 4; ++nt) bp[nt] = Wt + (size_t)(n0 + nt * 16 + lc) * DIM;

  v8f acc0[4] = {{}, {}, {}, {}};   // row-strip 0, 4 col tiles
  v8f acc1[4] = {{}, {}, {}, {}};   // row-strip 1

#pragma unroll
  for (int ks = 0; ks < DIM; ks += 32) {
    v16h a0 = ldfrag(arow0, ks, kb);
    v16h a1 = ldfrag(arow1, ks, kb);
#pragma unroll
    for (int nt = 0; nt < 4; ++nt) {
      v16h b = ldfrag(bp[nt], ks, kb);
      acc0[nt] = __builtin_amdgcn_wmma_f32_16x16x32_f16(false, a0, false, b,
                                                        (short)0, acc0[nt], false, false);
      acc1[nt] = __builtin_amdgcn_wmma_f32_16x16x32_f16(false, a1, false, b,
                                                        (short)0, acc1[nt], false, false);
    }
  }

  // epilogue: C layout lane = col l&15, rows (l>>4)*8 + r within each 16-row strip
  float bv[4];
#pragma unroll
  for (int nt = 0; nt < 4; ++nt) bv[nt] = bias[n0 + nt * 16 + lc];

  const int mb0 = m0 + kb;
  const int mb1 = m0 + 16 + kb;
#pragma unroll
  for (int r = 0; r < 8; ++r) {
    int ma = mb0 + r;
    if (ma < M) {
      float* crow = C + (size_t)ma * DIM;
#pragma unroll
      for (int nt = 0; nt < 4; ++nt) crow[n0 + nt * 16 + lc] = acc0[nt][r] + bv[nt];
    }
    int mbq = mb1 + r;
    if (mbq < M) {
      float* crow = C + (size_t)mbq * DIM;
#pragma unroll
      for (int nt = 0; nt < 4; ++nt) crow[n0 + nt * 16 + lc] = acc1[nt][r] + bv[nt];
    }
  }
}

// ---------------- edge scatter: agg[dst] += Wh[src] * inv_cnt[dst] ----------------
// one wave per edge; 8 floats per lane; f32 atomics resolve in the 192MB L2

__global__ __launch_bounds__(256) void k_scatter(const float* __restrict__ Wh,
                                                 const int* __restrict__ src,
                                                 const int* __restrict__ dst,
                                                 const float* __restrict__ inv,
                                                 float* __restrict__ out, int E) {
  int gw   = (blockIdx.x * 256 + threadIdx.x) >> 5;
  int lane = threadIdx.x & 31;
  if (gw >= E) return;
  int s = src[gw];
  int d = dst[gw];
  float w = inv[d];
  const float4* in4 = (const float4*)(Wh + (size_t)s * DIM);
  float4 v0 = in4[lane];
  float4 v1 = in4[lane + 32];
  float* o = out + (size_t)d * DIM + lane * 4;
  unsafeAtomicAdd(o + 0, v0.x * w);
  unsafeAtomicAdd(o + 1, v0.y * w);
  unsafeAtomicAdd(o + 2, v0.z * w);
  unsafeAtomicAdd(o + 3, v0.w * w);
  unsafeAtomicAdd(o + 128 + 0, v1.x * w);
  unsafeAtomicAdd(o + 128 + 1, v1.y * w);
  unsafeAtomicAdd(o + 128 + 2, v1.z * w);
  unsafeAtomicAdd(o + 128 + 3, v1.w * w);
}

// ---------------- host orchestration ----------------

extern "C" void kernel_launch(void* const* d_in, const int* in_sizes, int n_in,
                              void* d_out, int out_size, void* d_ws, size_t ws_size,
                              hipStream_t stream) {
  (void)n_in; (void)ws_size;
  const float* emb_ch = (const float*)d_in[0];
  const float* emb_ge = (const float*)d_in[1];
  const int M_ch = in_sizes[0] / DIM;
  const int M_ge = in_sizes[1] / DIM;
  const int maxM = (M_ch > M_ge) ? M_ch : M_ge;

  // etypes: 0 ch2ge, 1 ge2ch, 2 ch2ch, 3 ge2ge
  const int srcIsCh[4] = {1, 0, 1, 0};
  const int dstIsCh[4] = {0, 1, 1, 0};
  const float *W1[4], *B1[4], *W2[4], *B2[4];
  const int *SRC[4], *DST[4];
  int Eet[4];
  for (int e = 0; e < 4; ++e) {
    int b = 2 + 6 * e;
    W1[e] = (const float*)d_in[b + 0];
    B1[e] = (const float*)d_in[b + 1];
    W2[e] = (const float*)d_in[b + 2];
    B2[e] = (const float*)d_in[b + 3];
    SRC[e] = (const int*)d_in[b + 4];
    DST[e] = (const int*)d_in[b + 5];
    Eet[e] = in_sizes[b + 4];
  }

  // workspace carve-up
  char* p = (char*)d_ws;
  float*    agg_ch = (float*)p;    p += (size_t)M_ch * DIM * sizeof(float);
  float*    agg_ge = (float*)p;    p += (size_t)M_ge * DIM * sizeof(float);
  float*    Wh     = (float*)p;    p += (size_t)maxM * DIM * sizeof(float);
  _Float16* h16_ch = (_Float16*)p; p += (size_t)M_ch * DIM * sizeof(_Float16);
  _Float16* h16_ge = (_Float16*)p; p += (size_t)M_ge * DIM * sizeof(_Float16);
  _Float16* Wt[8];
  for (int i = 0; i < 8; ++i) { Wt[i] = (_Float16*)p; p += (size_t)DIM * DIM * sizeof(_Float16); }
  float* inv[4];
  for (int e = 0; e < 4; ++e) { inv[e] = (float*)p; p += (size_t)maxM * sizeof(float); }

  auto cdiv = [](int a, int b) { return (a + b - 1) / b; };

  // 1) zero accumulators and output
  k_zero<<<cdiv(M_ch * DIM, 256), 256, 0, stream>>>(agg_ch, M_ch * DIM);
  k_zero<<<cdiv(M_ge * DIM, 256), 256, 0, stream>>>(agg_ge, M_ge * DIM);
  k_zero<<<cdiv(out_size, 256), 256, 0, stream>>>((float*)d_out, out_size);

  // 2) per-etype reciprocal mean counts (edge lists are identical in both layers)
  for (int e = 0; e < 4; ++e) {
    int Md = dstIsCh[e] ? M_ch : M_ge;
    k_zero  <<<cdiv(Md, 256), 256, 0, stream>>>(inv[e], Md);
    k_hist  <<<cdiv(Eet[e], 256), 256, 0, stream>>>(DST[e], (int*)inv[e], Eet[e]);
    k_invert<<<cdiv(Md, 256), 256, 0, stream>>>(inv[e], Md);
  }

  // 3) transpose+convert weights to f16 column-major
  for (int e = 0; e < 4; ++e) {
    k_wtrans<<<DIM, 256, 0, stream>>>(W1[e], Wt[e]);
    k_wtrans<<<DIM, 256, 0, stream>>>(W2[e], Wt[4 + e]);
  }

  // 4) embeddings to f16
  k_cvt_f16<<<cdiv(M_ch * DIM, 256), 256, 0, stream>>>(emb_ch, h16_ch, M_ch * DIM);
  k_cvt_f16<<<cdiv(M_ge * DIM, 256), 256, 0, stream>>>(emb_ge, h16_ge, M_ge * DIM);

  // 5) layer 1: all 4 etypes
  for (int e = 0; e < 4; ++e) {
    const _Float16* A = srcIsCh[e] ? h16_ch : h16_ge;
    int Ms = srcIsCh[e] ? M_ch : M_ge;
    dim3 g(cdiv(Ms, 32), 4);
    k_gemm_wmma<<<g, 32, 0, stream>>>(A, Wt[e], B1[e], Wh, Ms);
    float* agg = dstIsCh[e] ? agg_ch : agg_ge;
    k_scatter<<<cdiv(Eet[e] * 32, 256), 256, 0, stream>>>(Wh, SRC[e], DST[e], inv[e], agg, Eet[e]);
  }

  // 6) leaky-relu + f16 convert for layer-2 inputs
  k_lrelu_cvt<<<cdiv(M_ch * DIM, 256), 256, 0, stream>>>(agg_ch, h16_ch, M_ch * DIM);
  k_lrelu_cvt<<<cdiv(M_ge * DIM, 256), 256, 0, stream>>>(agg_ge, h16_ge, M_ge * DIM);

  // 7) layer 2: only etypes whose dst is "chemical" contribute to the output
  for (int e = 1; e <= 2; ++e) {   // 1 = ge2ch, 2 = ch2ch
    const _Float16* A = srcIsCh[e] ? h16_ch : h16_ge;
    int Ms = srcIsCh[e] ? M_ch : M_ge;
    dim3 g(cdiv(Ms, 32), 4);
    k_gemm_wmma<<<g, 32, 0, stream>>>(A, Wt[4 + e], B2[e], Wh, Ms);
    k_scatter<<<cdiv(Eet[e] * 32, 256), 256, 0, stream>>>(Wh, SRC[e], DST[e], inv[e], (float*)d_out, Eet[e]);
  }
}
